// Sub_BidirectionalLSTM_27462020891201
// MI455X (gfx1250) — compile-verified, matrix operand order checked by reading, not run on hardware
//
#include <hip/hip_runtime.h>
#include <hip/hip_bf16.h>
#include <math.h>

// ---------------------------------------------------------------------------
// Bidirectional-LSTM ensemble for MI455X (gfx1250, wave32, WMMA bf16).
// ---------------------------------------------------------------------------

typedef __attribute__((ext_vector_type(16))) __bf16 v16bf;
typedef __attribute__((ext_vector_type(8)))  float  v8f;

union FragBF {
  v16bf v;
  uint4 q[2];
};

#define B_  64
#define T_  128
#define I_  512
#define H_  256
#define G_  1024   /* 4H */
#define O_  512
#define BT_ 8192   /* B*T */

// round-to-nearest-even f32 -> bf16 (bit pattern)
__device__ __forceinline__ unsigned short f2bf(float f) {
  unsigned int u = __float_as_uint(f);
  u += 0x7FFFu + ((u >> 16) & 1u);
  return (unsigned short)(u >> 16);
}

// Load a 16x32 bf16 A-fragment (or the mirrored 32x16 B-fragment built from a
// row-major W[N,K]) per the CDNA5 ISA VGPR layout:
//   lane r=lane&15 -> matrix row (A: M, B: N), g = (lane>=16) ? 8 : 0
//   vector elems 0..7  = K in [k0+g,    k0+g+8)   (one b128 load)
//   vector elems 8..15 = K in [k0+g+16, k0+g+24)  (one b128 load)
__device__ __forceinline__ v16bf load_frag(const unsigned short* base, int ld,
                                           int row0, int k0, int lane) {
  const int r = lane & 15;
  const int g = (lane & 16) ? 8 : 0;
  const unsigned short* p = base + (size_t)(row0 + r) * ld + (k0 + g);
  FragBF f;
  f.q[0] = *reinterpret_cast<const uint4*>(p);
  f.q[1] = *reinterpret_cast<const uint4*>(p + 16);
  return f.v;
}

__device__ __forceinline__ float sigm(float x) { return 1.0f / (1.0f + __expf(-x)); }

// Deterministic float atomic-max via the ordered-int encoding.
__device__ __forceinline__ void atom_max_f32(float* addr, float v) {
  if (v >= 0.0f) atomicMax(reinterpret_cast<int*>(addr), __float_as_int(v));
  else           atomicMin(reinterpret_cast<unsigned int*>(addr), __float_as_uint(v));
}

__global__ void cvt_f32_bf16_kernel(const float* __restrict__ in,
                                    unsigned short* __restrict__ out, int n) {
  int i = blockIdx.x * blockDim.x + threadIdx.x;
  if (i < n) out[i] = f2bf(in[i]);
}

__global__ void fill_neginf_kernel(int* __restrict__ out, int n) {
  int i = blockIdx.x * blockDim.x + threadIdx.x;
  if (i < n) out[i] = 0xFF800000; // -inf
}

// ---------------------------------------------------------------------------
// C[M,N] = A[M,K] @ W[N,K]^T + bias[N];  A,W bf16 row-major; C f32.
// Block = 8 waves; wave owns a 16x64 strip (4 N-tiles sharing one A-frag).
// ---------------------------------------------------------------------------
__global__ __launch_bounds__(256)
void gemm_bf16_kernel(const unsigned short* __restrict__ A,
                      const unsigned short* __restrict__ W,
                      const float* __restrict__ bias,
                      float* __restrict__ C, int M, int N, int K) {
  const int lane = threadIdx.x & 31;
  const int wave = threadIdx.x >> 5;
  const int m0 = blockIdx.x * 16;
  const int n0 = (blockIdx.y * 8 + wave) * 64;
  const v8f vz = {0.f,0.f,0.f,0.f,0.f,0.f,0.f,0.f};
  v8f acc[4] = {vz, vz, vz, vz};
  for (int k0 = 0; k0 < K; k0 += 32) {
    v16bf a = load_frag(A, K, m0, k0, lane);
#pragma unroll
    for (int q = 0; q < 4; ++q) {
      v16bf b = load_frag(W, K, n0 + q * 16, k0, lane);
      acc[q] = __builtin_amdgcn_wmma_f32_16x16x32_bf16(
          false, a, false, b, (short)0, acc[q], false, false);
    }
  }
  const int col = lane & 15;
  const int rb  = (lane & 16) ? 8 : 0;
#pragma unroll
  for (int q = 0; q < 4; ++q) {
    const int nc = n0 + q * 16 + col;
    const float bv = bias[nc];
#pragma unroll
    for (int v = 0; v < 8; ++v)
      C[(size_t)(m0 + rb + v) * N + nc] = acc[q][v] + bv;
  }
}

// ---------------------------------------------------------------------------
// Persistent LSTM scan. One block = (run, 16-row batch chunk).
// 16 waves; wave w owns gate columns [16w,16w+16) of ALL FOUR gates, so the
// nonlinearity + cell update is wave-local (c stays in accumulator registers).
// w_hh fragments are VGPR-resident across the whole time loop; h is bf16 in
// LDS and re-read as WMMA A-fragments each step. xp loads for the current
// step are issued BEFORE the WMMA loop (latency overlapped with the matrix
// work), and the next step's xp lines are prefetched (global_prefetch_b8).
//
// phase 1 grid.x = 52 runs:
//   0            : full fwd  -> store Hf, atomic-max NI[ch 0..H) for t<25
//   1            : full bwd  -> store Hb, atomic-max NI[ch H..2H) for t>=1
//   2..26  (i=run-1)  : prefix bwd over [0,i)   -> atomic-max NI[ch H..2H)
//   27..51 (j=run-26) : suffix fwd over [j,T)   -> atomic-max NI[ch 0..H)
// phase 2 grid.x = 2: fwd/bwd over maxed input -> store R[:, :, 0:H / H:2H].
// ---------------------------------------------------------------------------
__global__ __launch_bounds__(512)
void lstm_scan_kernel(const float* __restrict__ xp_f, const float* __restrict__ xp_b,
                      const unsigned short* __restrict__ whh_f,
                      const unsigned short* __restrict__ whh_b,
                      float* Hf, float* Hb, float* NI, float* R, int phase) {
  const int lane  = threadIdx.x & 31;
  const int w     = threadIdx.x >> 5;     // 0..15
  const int chunk = blockIdx.y;           // 0..3 (16 batch rows each)

  bool fwd; int t0, t1;
  const float* xp; const unsigned short* whh;
  float* st = nullptr; int st_ld = 0, st_ch = 0;
  bool doat = false; int ach = 0, alo = 0, ahi = 0;

  if (phase == 1) {
    const int run = blockIdx.x;
    if (run == 0)       { fwd = true;  t0 = 0; t1 = T_; xp = xp_f; whh = whh_f;
                          st = Hf; st_ld = H_; st_ch = 0;
                          doat = true; ach = 0;  alo = 0; ahi = 25; }
    else if (run == 1)  { fwd = false; t0 = 0; t1 = T_; xp = xp_b; whh = whh_b;
                          st = Hb; st_ld = H_; st_ch = 0;
                          doat = true; ach = H_; alo = 1; ahi = T_; }
    else if (run <= 26) { const int i = run - 1;  fwd = false; t0 = 0; t1 = i;
                          xp = xp_b; whh = whh_b;
                          doat = true; ach = H_; alo = 0; ahi = T_; }
    else                { const int j = run - 26; fwd = true;  t0 = j; t1 = T_;
                          xp = xp_f; whh = whh_f;
                          doat = true; ach = 0;  alo = 0; ahi = T_; }
  } else {
    fwd = (blockIdx.x == 0);
    t0 = 0; t1 = T_;
    if (fwd) { xp = xp_f; whh = whh_f; st = R; st_ld = 2 * H_; st_ch = 0;  }
    else     { xp = xp_b; whh = whh_b; st = R; st_ld = 2 * H_; st_ch = H_; }
  }

  // h state, bf16. Row stride 272 (*2B = 544 = 34*16) keeps every fragment
  // b128 load 16-byte aligned.
  __shared__ __align__(16) unsigned short h_lds[16][272];
  for (int i = threadIdx.x; i < 16 * 272; i += blockDim.x)
    (&h_lds[0][0])[i] = 0;

  // Preload w_hh B-fragments: gate q, columns 16w..16w+15, K = 0..255.
  v16bf bfrag[4][8];
#pragma unroll
  for (int q = 0; q < 4; ++q)
#pragma unroll
    for (int kk = 0; kk < 8; ++kk)
      bfrag[q][kk] = load_frag(whh, H_, q * H_ + w * 16, kk * 32, lane);

  const v8f vz = {0.f,0.f,0.f,0.f,0.f,0.f,0.f,0.f};
  v8f cstate = vz;

  const int col  = lane & 15;
  const int rb   = (lane & 16) ? 8 : 0;
  const int hidx = w * 16 + col;          // hidden unit owned by (wave,lane)
  __syncthreads();

  const int steps = t1 - t0;
  const int tstep = fwd ? 1 : -1;
  for (int tt = 0; tt < steps; ++tt) {
    const int t = fwd ? (t0 + tt) : (t1 - 1 - tt);

    // ---- Issue this step's xp loads up front: their latency is hidden
    //      behind the 8 LDS fragment loads + 32 WMMAs below.
    float xg[4][8];
#pragma unroll
    for (int v = 0; v < 8; ++v) {
      const int brow = chunk * 16 + rb + v;
      const size_t xb = ((size_t)brow * T_ + t) * G_;
#pragma unroll
      for (int q = 0; q < 4; ++q)
        xg[q][v] = xp[xb + q * H_ + hidx];
    }

    // ---- Prefetch the next step's xp lines (global_prefetch_b8); the walk
    //      is fully deterministic so these land in-cache ahead of use.
    if (tt + 1 < steps) {
      const int tn = t + tstep;
#pragma unroll
      for (int v = 0; v < 8; v += 4) {
        const int brow = chunk * 16 + rb + v;
        const size_t xb = ((size_t)brow * T_ + tn) * G_;
#pragma unroll
        for (int q = 0; q < 4; ++q)
          __builtin_prefetch(&xp[xb + q * H_ + hidx], 0, 1);
      }
    }

    // ---- Recurrence GEMM: g += h_prev @ w_hh^T (per-wave 16x64 strip).
    v8f acc0 = vz, acc1 = vz, acc2 = vz, acc3 = vz;
#pragma unroll
    for (int kk = 0; kk < 8; ++kk) {
      v16bf a = load_frag(&h_lds[0][0], 272, 0, kk * 32, lane);
      acc0 = __builtin_amdgcn_wmma_f32_16x16x32_bf16(false, a, false, bfrag[0][kk], (short)0, acc0, false, false);
      acc1 = __builtin_amdgcn_wmma_f32_16x16x32_bf16(false, a, false, bfrag[1][kk], (short)0, acc1, false, false);
      acc2 = __builtin_amdgcn_wmma_f32_16x16x32_bf16(false, a, false, bfrag[2][kk], (short)0, acc2, false, false);
      acc3 = __builtin_amdgcn_wmma_f32_16x16x32_bf16(false, a, false, bfrag[3][kk], (short)0, acc3, false, false);
    }
    __syncthreads();   // every wave done reading h_lds for this step

    // ---- Wave-local gate fusion + cell update (c stays in registers).
    float hv[8];
#pragma unroll
    for (int v = 0; v < 8; ++v) {
      const float gi = sigm (xg[0][v] + acc0[v]);
      const float gf = sigm (xg[1][v] + acc1[v]);
      const float gg = tanhf(xg[2][v] + acc2[v]);
      const float go = sigm (xg[3][v] + acc3[v]);
      const float c  = gf * cstate[v] + gi * gg;
      cstate[v] = c;
      hv[v] = go * tanhf(c);
    }

#pragma unroll
    for (int v = 0; v < 8; ++v) {
      h_lds[rb + v][hidx] = f2bf(hv[v]);
      const int brow = chunk * 16 + rb + v;
      const size_t rbase = (size_t)brow * T_ + t;
      if (st) st[rbase * st_ld + st_ch + hidx] = hv[v];
      if (doat && t >= alo && t < ahi)
        atom_max_f32(&NI[rbase * (2 * H_) + ach + hidx], hv[v]);
    }
    __syncthreads();   // h_lds ready for next step
  }
}

// ---------------------------------------------------------------------------
extern "C" void kernel_launch(void* const* d_in, const int* in_sizes, int n_in,
                              void* d_out, int out_size, void* d_ws, size_t ws_size,
                              hipStream_t stream) {
  (void)in_sizes; (void)n_in; (void)out_size; (void)ws_size;

  const float* x    = (const float*)d_in[0];
  const float* wihf = (const float*)d_in[1];
  const float* whhf = (const float*)d_in[2];
  const float* bf   = (const float*)d_in[3];
  const float* wihb = (const float*)d_in[4];
  const float* whhb = (const float*)d_in[5];
  const float* bb   = (const float*)d_in[6];
  const float* wlin = (const float*)d_in[7];
  const float* blin = (const float*)d_in[8];

  char* p = (char*)d_ws;
  auto take = [&](size_t bytes) -> char* {
    char* r = p; p += (bytes + 255) & ~(size_t)255; return r;
  };

  unsigned short* xbf    = (unsigned short*)take((size_t)BT_ * I_ * 2);
  unsigned short* wihf_b = (unsigned short*)take((size_t)G_ * I_ * 2);
  unsigned short* wihb_b = (unsigned short*)take((size_t)G_ * I_ * 2);
  unsigned short* whhf_b = (unsigned short*)take((size_t)G_ * H_ * 2);
  unsigned short* whhb_b = (unsigned short*)take((size_t)G_ * H_ * 2);
  unsigned short* wlin_b = (unsigned short*)take((size_t)O_ * (2 * H_) * 2);
  float* xpf = (float*)take((size_t)BT_ * G_ * 4);   // reused for phase-2 xp
  float* xpb = (float*)take((size_t)BT_ * G_ * 4);
  float* Hf  = (float*)take((size_t)BT_ * H_ * 4);
  float* Hb  = (float*)take((size_t)BT_ * H_ * 4);
  float* NI  = (float*)take((size_t)BT_ * (2 * H_) * 4);
  unsigned short* NIbf = (unsigned short*)take((size_t)BT_ * (2 * H_) * 2);
  float* R = Hf;                 // alias: Hf+Hb contiguous, dead after phase 1
  unsigned short* Rbf = NIbf;    // alias: NIbf dead after the xp2 GEMMs

  auto cvt = [&](const float* in, unsigned short* out, int n) {
    cvt_f32_bf16_kernel<<<(n + 255) / 256, 256, 0, stream>>>(in, out, n);
  };

  // 1) bf16 copies of activations + weights
  cvt(x,    xbf,    BT_ * I_);
  cvt(wihf, wihf_b, G_ * I_);
  cvt(wihb, wihb_b, G_ * I_);
  cvt(whhf, whhf_b, G_ * H_);
  cvt(whhb, whhb_b, G_ * H_);
  cvt(wlin, wlin_b, O_ * 2 * H_);

  // 2) combine accumulator := -inf
  fill_neginf_kernel<<<(BT_ * 2 * H_ + 255) / 256, 256, 0, stream>>>(
      (int*)NI, BT_ * 2 * H_);

  // 3) shared input projections (feed ALL 52 phase-1 runs)
  dim3 g1(BT_ / 16, G_ / 512);
  gemm_bf16_kernel<<<g1, 256, 0, stream>>>(xbf, wihf_b, bf, xpf, BT_, G_, I_);
  gemm_bf16_kernel<<<g1, 256, 0, stream>>>(xbf, wihb_b, bb, xpb, BT_, G_, I_);

  // 4) 52 independent recurrences (full fwd/bwd + 25 prefixes + 25 suffixes),
  //    max-merged into NI on the fly
  lstm_scan_kernel<<<dim3(52, 4), 512, 0, stream>>>(
      xpf, xpb, whhf_b, whhb_b, Hf, Hb, NI, R, 1);

  // 5) second bidirectional pass over the maxed input
  cvt(NI, NIbf, BT_ * 2 * H_);
  gemm_bf16_kernel<<<g1, 256, 0, stream>>>(NIbf, wihf_b, bf, xpf, BT_, G_, 2 * H_);
  gemm_bf16_kernel<<<g1, 256, 0, stream>>>(NIbf, wihb_b, bb, xpb, BT_, G_, 2 * H_);
  lstm_scan_kernel<<<dim3(2, 4), 512, 0, stream>>>(
      xpf, xpb, whhf_b, whhb_b, Hf, Hb, NI, R, 2);

  // 6) final linear [2H -> O]
  cvt(R, Rbf, BT_ * 2 * H_);
  dim3 g2(BT_ / 16, O_ / 512);
  gemm_bf16_kernel<<<g2, 256, 0, stream>>>(
      Rbf, wlin_b, blin, (float*)d_out, BT_, O_, 2 * H_);
}
